// Attention_67388036874492
// MI455X (gfx1250) — matrix operand order repaired and checked by reading, not compiled, not run on hardware
//
#include <hip/hip_runtime.h>

typedef __attribute__((ext_vector_type(16))) __bf16 v16bf;
typedef __attribute__((ext_vector_type(8)))  __bf16 bf16x8;
typedef __attribute__((ext_vector_type(8)))  float  v8f;
typedef __attribute__((ext_vector_type(4)))  unsigned int u32x4;
typedef __attribute__((ext_vector_type(4)))  int i32x4;
typedef __attribute__((ext_vector_type(8)))  int i32x8;

static __device__ __forceinline__ __bf16 f2bf(float f) {
  union { float f; unsigned u; } v; v.f = f;
  unsigned r = v.u + 0x7FFFu + ((v.u >> 16) & 1u);   // round-to-nearest-even
  union { unsigned short u; __bf16 b; } o; o.u = (unsigned short)(r >> 16);
  return o.b;
}

// Load a 16-element bf16 fragment from two 16-byte groups.
static __device__ __forceinline__ v16bf load_frag2(const __bf16* g0p, const __bf16* g1p) {
  bf16x8 g0 = *(const bf16x8*)g0p;
  bf16x8 g1 = *(const bf16x8*)g1p;
  v16bf r;
#pragma unroll
  for (int i = 0; i < 8; ++i) { r[i] = g0[i]; r[8 + i] = g1[i]; }
  return r;
}

// ---------------------------------------------------------------------------
// TDM: issue a 2-D tile load (global -> LDS). Builds D# groups 0/1 per CDNA5
// CDNA5 ISA. data_size fixed at 2 bytes (bf16).
// Tile iterates X (tile_d0, contiguous) fastest then Y (tile_d1) with row
// stride stride0 (in elements); LDS receives the tile packed row-major.
// ---------------------------------------------------------------------------
static __device__ __forceinline__ void tdm_load_2d(unsigned lds_off,
                                                   unsigned long long gaddr,
                                                   unsigned tensor_d0, unsigned tensor_d1,
                                                   unsigned tile_d0, unsigned tile_d1,
                                                   unsigned long long stride0) {
  u32x4 g0;
  g0[0] = 1u;                                              // count=1 (valid), user mode
  g0[1] = lds_off;                                         // lds_addr (bytes)
  g0[2] = (unsigned)(gaddr & 0xffffffffu);                 // global_addr[31:0]
  g0[3] = (unsigned)((gaddr >> 32) & 0x1ffffffu) | (2u << 30); // global_addr[56:32] | type=2
  i32x8 g1;
  g1[0] = (int)(1u << 16);                                 // data_size=1 (2 bytes)
  g1[1] = (int)((tensor_d0 & 0xffffu) << 16);              // tensor_dim0[15:0] @bit48
  g1[2] = (int)(((tensor_d0 >> 16) & 0xffffu) | ((tensor_d1 & 0xffffu) << 16));
  g1[3] = (int)(((tensor_d1 >> 16) & 0xffffu) | ((tile_d0 & 0xffffu) << 16));
  g1[4] = (int)(tile_d1 & 0xffffu);                        // tile_dim1 | tile_dim2=0
  g1[5] = (int)(unsigned)(stride0 & 0xffffffffu);          // tensor_dim0_stride[31:0]
  g1[6] = (int)(unsigned)((stride0 >> 32) & 0xffffu);      // stride0[47:32] | stride1[15:0]=0
  g1[7] = 0;
  i32x4 gz = {0, 0, 0, 0};
#if __clang_major__ >= 23
  i32x8 gz8 = {0, 0, 0, 0, 0, 0, 0, 0};
  __builtin_amdgcn_tensor_load_to_lds(g0, g1, gz, gz, gz8, 0);
#else
  __builtin_amdgcn_tensor_load_to_lds(g0, g1, gz, gz, 0);
#endif
}

// ---------------------------------------------------------------------------
// Prep kernels: fp32 -> bf16 (weights) and transpose+convert for x.
// ---------------------------------------------------------------------------
__global__ void cvt_bf16(const float* __restrict__ in, __bf16* __restrict__ out, int n) {
  int i = blockIdx.x * blockDim.x + threadIdx.x;
  if (i < n) out[i] = f2bf(in[i]);
}

// x: [256][4096] fp32 (channel-major) -> xt: [4096][256] bf16 (position-major)
__global__ void xpose_cvt(const float* __restrict__ x, __bf16* __restrict__ xt) {
  int idx = blockIdx.x * blockDim.x + threadIdx.x;   // 0..1048575
  int c = idx & 255, p = idx >> 8;
  xt[idx] = f2bf(x[(size_t)c * 4096 + p]);
}

// ---------------------------------------------------------------------------
// Generic WMMA GEMM: C[MxN] = A[MxK] * Bt[NxK]^T, epilogue val*scale[o]+bias[o].
// One wave per workgroup; tile = 16 rows x 64 cols (4 N-tiles share the A frag).
// A-frag (16x32 bf16): lane L+16s holds row L, k-groups at 8s and 16+8s.
// B-frag (32x16 bf16): lane L+16s holds col L, k-groups at 8s and 16+8s.
// C/D (16x16 f32): lane L+16s, elem j -> [m=j+8s][n=L].
// mode 0: bf16 out at [h=o>>5][p][d=o&31] (stride 32)  -- Q/K layout
// mode 1: bf16 out at [o][p] (stride 4096)             -- V layout
// mode 2: f32  out at [o][p] (stride 4096)             -- final output
// ---------------------------------------------------------------------------
__global__ void gemm_wmma(const __bf16* __restrict__ A, const __bf16* __restrict__ Bt,
                          const float* __restrict__ scale, const float* __restrict__ bias,
                          int K, void* __restrict__ out, int mode) {
  const int lane = threadIdx.x;
  const int L = lane & 15, s = lane >> 4;
  const int m0 = blockIdx.y * 16;
  const int n0 = blockIdx.x * 64;
  const __bf16* arow = A + (size_t)(m0 + L) * K;

  v8f acc[4] = {};
  for (int kb = 0; kb < K; kb += 32) {
    v16bf af = load_frag2(arow + kb + 8 * s, arow + kb + 16 + 8 * s);
#pragma unroll
    for (int t = 0; t < 4; ++t) {
      const __bf16* bcol = Bt + (size_t)(n0 + t * 16 + L) * K + kb + 8 * s;
      v16bf bf_ = load_frag2(bcol, bcol + 16);
      acc[t] = __builtin_amdgcn_wmma_f32_16x16x32_bf16(false, af, false, bf_,
                                                       (short)0, acc[t], false, false);
    }
  }

  float sc[8], bi[8];
#pragma unroll
  for (int j = 0; j < 8; ++j) { int o = m0 + j + 8 * s; sc[j] = scale[o]; bi[j] = bias[o]; }

#pragma unroll
  for (int t = 0; t < 4; ++t) {
    const int p = n0 + t * 16 + L;
#pragma unroll
    for (int j = 0; j < 8; ++j) {
      const int o = m0 + j + 8 * s;
      float val = acc[t][j] * sc[j] + bi[j];
      if (mode == 0) {
        const int hh = o >> 5, d = o & 31;
        ((__bf16*)out)[(((size_t)hh * 4096) + p) * 32 + d] = f2bf(val);
      } else if (mode == 1) {
        ((__bf16*)out)[(size_t)o * 4096 + p] = f2bf(val);
      } else {
        ((float*)out)[(size_t)o * 4096 + p] = val;
      }
    }
  }
}

// ---------------------------------------------------------------------------
// Flash attention: 256 threads = 8 waves per workgroup; each wave owns 16
// queries (128 queries/WG). Per 32-key step the TDM (issued by wave 0,
// double-buffered, overlapped via s_wait_tensorcnt<=2) stages the shared
// 32x32 K tile and strided 128x32 V tile into LDS; all waves build WMMA
// fragments from LDS. Online-softmax state lives replicated per 16-lane half.
// Qb/Kb: [8][4096][32] bf16, Vb: [8][128][4096] bf16, Ob: [4096][1024] bf16.
// ---------------------------------------------------------------------------
__global__ void __launch_bounds__(256) flash_attn(
    const __bf16* __restrict__ Qb, const __bf16* __restrict__ Kb,
    const __bf16* __restrict__ Vb, __bf16* __restrict__ Ob) {
  __shared__ __bf16 Kt[2][32 * 32];    // [key][d]   2 KB each
  __shared__ __bf16 Vt[2][128 * 32];   // [dv][key]  8 KB each
  __shared__ __bf16 Pl[8][16 * 32];    // per-wave P tile

  const int h  = blockIdx.y;
  const int w  = threadIdx.x >> 5;               // wave id 0..7
  const int lane = threadIdx.x & 31, L = lane & 15, s = lane >> 4;
  const int q0 = blockIdx.x * 128 + w * 16;

  const __bf16* Qh = Qb + (size_t)h * 4096 * 32;
  const unsigned long long Kg = (unsigned long long)(size_t)(Kb + (size_t)h * 4096 * 32);
  const unsigned long long Vg = (unsigned long long)(size_t)(Vb + (size_t)h * 128 * 4096);
  const unsigned koff[2] = { (unsigned)(size_t)(void*)&Kt[0][0], (unsigned)(size_t)(void*)&Kt[1][0] };
  const unsigned voff[2] = { (unsigned)(size_t)(void*)&Vt[0][0], (unsigned)(size_t)(void*)&Vt[1][0] };

  const __bf16* qrow = Qh + (size_t)(q0 + L) * 32;
  v16bf qf = load_frag2(qrow + 8 * s, qrow + 16 + 8 * s);

  v8f oacc[8] = {};
  float rmax[8], rsum[8];
#pragma unroll
  for (int j = 0; j < 8; ++j) { rmax[j] = -3.0e38f; rsum[j] = 0.f; }

  // Prologue: stage the first K/V tiles into buffer 0.
  if (w == 0) {
    tdm_load_2d(koff[0], Kg, 32, 4096, 32, 32, 32);         // K rows 0..31
    tdm_load_2d(voff[0], Vg, 4096, 128, 32, 128, 4096);     // V cols 0..31 (strided)
  }

  int cur = 0;
  for (int kb = 0; kb < 4096; kb += 32) {
    if (w == 0) {
      if (kb + 32 < 4096) {
        tdm_load_2d(koff[cur ^ 1], Kg + (unsigned long long)(kb + 32) * 64,
                    32, 4096, 32, 32, 32);
        tdm_load_2d(voff[cur ^ 1], Vg + (unsigned long long)(kb + 32) * 2,
                    4096, 128, 32, 128, 4096);
        __builtin_amdgcn_s_wait_tensorcnt(2);   // current tiles done, next in flight
      } else {
        __builtin_amdgcn_s_wait_tensorcnt(0);
      }
    }
    __syncthreads();                            // tiles[cur] visible to all waves

    const __bf16* Kl = &Kt[cur][0];
    const __bf16* Vl = &Vt[cur][0];

    // S = Q * K^T for this wave's 16 queries (reference applies no scale).
    const __bf16* k0 = Kl + L * 32 + 8 * s;
    const __bf16* k1 = Kl + (16 + L) * 32 + 8 * s;
    v16bf kf0 = load_frag2(k0, k0 + 16);
    v16bf kf1 = load_frag2(k1, k1 + 16);
    v8f z = {};
    v8f s0 = __builtin_amdgcn_wmma_f32_16x16x32_bf16(false, qf, false, kf0, (short)0, z, false, false);
    v8f s1 = __builtin_amdgcn_wmma_f32_16x16x32_bf16(false, qf, false, kf1, (short)0, z, false, false);

    __bf16* Pw = &Pl[w][0];
    float corr[8];
#pragma unroll
    for (int j = 0; j < 8; ++j) {
      float mx = fmaxf(s0[j], s1[j]);
      mx = fmaxf(mx, __shfl_xor(mx, 1));
      mx = fmaxf(mx, __shfl_xor(mx, 2));
      mx = fmaxf(mx, __shfl_xor(mx, 4));
      mx = fmaxf(mx, __shfl_xor(mx, 8));
      float nm = fmaxf(rmax[j], mx);
      corr[j] = __expf(rmax[j] - nm);
      rmax[j] = nm;
      float p0 = __expf(s0[j] - nm);
      float p1 = __expf(s1[j] - nm);
      Pw[(j + 8 * s) * 32 + L]      = f2bf(p0);   // row m=j+8s, col = key-in-block
      Pw[(j + 8 * s) * 32 + 16 + L] = f2bf(p1);
      float rs = p0 + p1;
      rs += __shfl_xor(rs, 1);
      rs += __shfl_xor(rs, 2);
      rs += __shfl_xor(rs, 4);
      rs += __shfl_xor(rs, 8);
      rsum[j] = rsum[j] * corr[j] + rs;
    }

    // rescale existing accumulators by per-row correction
#pragma unroll
    for (int t = 0; t < 8; ++t)
#pragma unroll
      for (int j = 0; j < 8; ++j) oacc[t][j] *= corr[j];

    // P tile is per-wave private; same-wave LDS ops stay in order.
    const __bf16* prow = Pw + L * 32;
    v16bf pf = load_frag2(prow + 8 * s, prow + 16 + 8 * s);

    // O += P * V over the 128-wide value dim (8 N-tiles) from LDS.
#pragma unroll
    for (int t = 0; t < 8; ++t) {
      const __bf16* vp = Vl + (t * 16 + L) * 32 + 8 * s;
      v16bf vf = load_frag2(vp, vp + 16);
      oacc[t] = __builtin_amdgcn_wmma_f32_16x16x32_bf16(false, pf, false, vf,
                                                        (short)0, oacc[t], false, false);
    }

    __syncthreads();   // everyone done with tiles[cur] before TDM overwrites it
    cur ^= 1;
  }

  // Finalize and store as Ob[p][c] bf16 (c = h*128 + dv), ready for out-proj.
#pragma unroll
  for (int t = 0; t < 8; ++t) {
    const int c = h * 128 + t * 16 + L;
#pragma unroll
    for (int j = 0; j < 8; ++j) {
      const int q = q0 + j + 8 * s;
      Ob[(size_t)q * 1024 + c] = f2bf(oacc[t][j] / rsum[j]);
    }
  }
}

// ---------------------------------------------------------------------------
extern "C" void kernel_launch(void* const* d_in, const int* in_sizes, int n_in,
                              void* d_out, int out_size, void* d_ws, size_t ws_size,
                              hipStream_t stream) {
  const float* x  = (const float*)d_in[0];
  const float* wq = (const float*)d_in[1];
  const float* sq = (const float*)d_in[2];
  const float* tq = (const float*)d_in[3];
  const float* wk = (const float*)d_in[4];
  const float* sk = (const float*)d_in[5];
  const float* tk = (const float*)d_in[6];
  const float* wv = (const float*)d_in[7];
  const float* sv = (const float*)d_in[8];
  const float* tv = (const float*)d_in[9];
  const float* wp = (const float*)d_in[10];
  const float* sp = (const float*)d_in[11];
  const float* tp = (const float*)d_in[12];

  char* ws = (char*)d_ws;
  auto carve = [&](size_t bytes) -> char* {
    char* p = ws; ws += (bytes + 255) & ~(size_t)255; return p;
  };
  __bf16* Xt  = (__bf16*)carve(4096ull * 256 * 2);   // x transposed [p][c]
  __bf16* Wqb = (__bf16*)carve(256ull * 256 * 2);
  __bf16* Wkb = (__bf16*)carve(256ull * 256 * 2);
  __bf16* Wvb = (__bf16*)carve(1024ull * 256 * 2);
  __bf16* Wpb = (__bf16*)carve(256ull * 1024 * 2);
  __bf16* Qb  = (__bf16*)carve(8ull * 4096 * 32 * 2);   // [h][n][32]
  __bf16* Kb  = (__bf16*)carve(8ull * 4096 * 32 * 2);   // [h][n][32]
  __bf16* Vb  = (__bf16*)carve(8ull * 128 * 4096 * 2);  // [h][128][n]
  __bf16* Ob  = (__bf16*)carve(4096ull * 1024 * 2);     // [p][c]

  xpose_cvt<<<4096, 256, 0, stream>>>(x, Xt);
  cvt_bf16<<<(65536 + 255) / 256, 256, 0, stream>>>(wq, Wqb, 65536);
  cvt_bf16<<<(65536 + 255) / 256, 256, 0, stream>>>(wk, Wkb, 65536);
  cvt_bf16<<<(262144 + 255) / 256, 256, 0, stream>>>(wv, Wvb, 262144);
  cvt_bf16<<<(262144 + 255) / 256, 256, 0, stream>>>(wp, Wpb, 262144);

  // Q, K projections: M=256, N=4096, K=256
  gemm_wmma<<<dim3(64, 16), 32, 0, stream>>>(Wqb, Xt, sq, tq, 256, (void*)Qb, 0);
  gemm_wmma<<<dim3(64, 16), 32, 0, stream>>>(Wkb, Xt, sk, tk, 256, (void*)Kb, 0);
  // V projection: M=1024
  gemm_wmma<<<dim3(64, 64), 32, 0, stream>>>(Wvb, Xt, sv, tv, 256, (void*)Vb, 1);

  // Flash attention: 32 query blocks (128 queries each) x 8 heads
  flash_attn<<<dim3(32, 8), 256, 0, stream>>>(Qb, Kb, Vb, Ob);

  // Output projection: M=256, N=4096, K=1024 -> fp32 d_out
  gemm_wmma<<<dim3(64, 16), 32, 0, stream>>>(Wpb, Ob, sp, tp, 1024, d_out, 2);
}
